// DiscreteActorCriticRNN_73658689126553
// MI455X (gfx1250) — compile-verified, matrix operand order checked by reading, not run on hardware
//
#include <hip/hip_runtime.h>

#define T_   1024
#define B_   512
#define OBS_ 64
#define H_   128
#define A_   16
#define BT_  16      // batch rows per workgroup
#define NTHR 256     // 8 waves (wave32)

typedef __attribute__((ext_vector_type(16))) __bf16 v16bf;
typedef __attribute__((ext_vector_type(8)))  float  v8f;

union Frag { v16bf v; unsigned int u[8]; };

__device__ __forceinline__ unsigned short f2bf(float f) {
    return __builtin_bit_cast(unsigned short, (__bf16)f);   // native bf16 convert (RNE)
}
__device__ __forceinline__ unsigned int pack2bf(float a, float b) {
    return (unsigned int)f2bf(a) | ((unsigned int)f2bf(b) << 16);
}
__device__ __forceinline__ float sigmoidf_(float x) { return 1.0f / (1.0f + __expf(-x)); }

__device__ __forceinline__ v8f wmma_bf16(const Frag& a, const Frag& b, v8f c) {
    return __builtin_amdgcn_wmma_f32_16x16x32_bf16(false, a.v, false, b.v, (short)0, c, false, false);
}

// ---- LDS layout (bytes). Total 310912 <= 320KB WGP LDS ----
#define OFFB_WEMB   0u        // uint[32*128]   (OBS/2 x H   kpair-packed bf16)
#define OFFB_WI     16384u    // uint[64*384]
#define OFFB_WH     114688u   // uint[64*384]
#define OFFB_WA1    212992u   // uint[64*128]
#define OFFB_WA2    245760u   // uint[64*16]
#define OFFB_WC1    249856u   // uint[64*128]
#define OFFB_WC2    282624u   // uint[64*16]  (N padded 1->16)
#define OFFB_BIAS   286720u   // floats: bemb128, bi384, bhn128, ba1 128, ba2 16, bc1 128, bc2 16
#define OFFB_HBF    290432u   // ushort[16*128]  y_t / h_t in bf16 (A-fragments)
#define OFFB_EMB    294528u   // ushort[16*128]
#define OFFB_OBS    298624u   // ushort[16*64]
#define OFFB_A1     300672u   // ushort[16*128]
#define OFFB_C1     304768u   // ushort[16*128]
#define OFFB_DMASK  308864u   // ushort[1024]  per-timestep 16-bit done mask for this block
#define SMEM_BYTES  310912u

// A-fragment (16x32 bf16): lanes 0-15 rows M=0..15 hold K {0..7,16..23}; lanes 16-31 +8.
// B-fragment: kpair-packed uint [K/2][N]; N = n, mirrored K pattern.

// single-output GEMM tile (emb / logits / value)
__device__ __forceinline__ v8f gemm1(const unsigned short* __restrict__ Abase, int ldk, int K,
                                     const unsigned int* __restrict__ Bp, int N, int n0,
                                     int lane, v8f acc) {
    const int m  = lane & 15;
    const int kb = (lane >> 4) << 3;
    const int n  = n0 + m;
    for (int k0 = 0; k0 < K; k0 += 32) {
        Frag a, b;
#pragma unroll
        for (int i = 0; i < 8; ++i) {
            const int k = (i < 4) ? (2 * i) : (16 + 2 * (i - 4));
            a.u[i] = *(const unsigned int*)(Abase + m * ldk + k0 + kb + k);
            b.u[i] = Bp[((k0 + kb + k) >> 1) * N + n];
        }
        acc = wmma_bf16(a, b, acc);
    }
    return acc;
}

// three outputs sharing one A tile: B columns n, n+128, n+256 of a [K/2][384] matrix
__device__ __forceinline__ void gemm3(const unsigned short* __restrict__ Abase, int ldk,
                                      const unsigned int* __restrict__ Bp, int n,
                                      int lane, v8f& c0, v8f& c1, v8f& c2) {
    const int m  = lane & 15;
    const int kb = (lane >> 4) << 3;
#pragma unroll
    for (int k0 = 0; k0 < H_; k0 += 32) {
        Frag a, b0, b1, b2;
#pragma unroll
        for (int i = 0; i < 8; ++i) {
            const int k = (i < 4) ? (2 * i) : (16 + 2 * (i - 4));
            a.u[i] = *(const unsigned int*)(Abase + m * ldk + k0 + kb + k);
            const unsigned int* bp = Bp + ((k0 + kb + k) >> 1) * (3 * H_) + n;
            b0.u[i] = bp[0];
            b1.u[i] = bp[H_];
            b2.u[i] = bp[2 * H_];
        }
        c0 = wmma_bf16(a, b0, c0);
        c1 = wmma_bf16(a, b1, c1);
        c2 = wmma_bf16(a, b2, c2);
    }
}

// two outputs sharing one A tile, two different B matrices (same N=128, same n0)
__device__ __forceinline__ void gemm2(const unsigned short* __restrict__ Abase, int ldk,
                                      const unsigned int* __restrict__ Bp0,
                                      const unsigned int* __restrict__ Bp1,
                                      int n, int lane, v8f& c0, v8f& c1) {
    const int m  = lane & 15;
    const int kb = (lane >> 4) << 3;
#pragma unroll
    for (int k0 = 0; k0 < H_; k0 += 32) {
        Frag a, b0, b1;
#pragma unroll
        for (int i = 0; i < 8; ++i) {
            const int k = (i < 4) ? (2 * i) : (16 + 2 * (i - 4));
            a.u[i] = *(const unsigned int*)(Abase + m * ldk + k0 + kb + k);
            const int kp = (k0 + kb + k) >> 1;
            b0.u[i] = Bp0[kp * H_ + n];
            b1.u[i] = Bp1[kp * H_ + n];
        }
        c0 = wmma_bf16(a, b0, c0);
        c1 = wmma_bf16(a, b1, c1);
    }
}

__device__ __forceinline__ void stage_packed(const float* g, unsigned int* d,
                                             int K, int N, int tid) {
    const int total = (K >> 1) * N;
    for (int e = tid; e < total; e += NTHR) {
        const int kp = e / N, n = e - kp * N;
        d[e] = pack2bf(g[(2 * kp) * N + n], g[(2 * kp + 1) * N + n]);
    }
}

__global__ __launch_bounds__(NTHR)
void actor_critic_gru_fused(const float* __restrict__ hidden0,
                            const float* __restrict__ obs,
                            const int*   __restrict__ dones,
                            const float* __restrict__ W_emb, const float* __restrict__ b_emb,
                            const float* __restrict__ Wi,    const float* __restrict__ bi,
                            const float* __restrict__ Wh,    const float* __restrict__ bh_n,
                            const float* __restrict__ Wa1,   const float* __restrict__ ba1,
                            const float* __restrict__ Wa2,   const float* __restrict__ ba2,
                            const float* __restrict__ Wc1,   const float* __restrict__ bc1,
                            const float* __restrict__ Wc2,   const float* __restrict__ bc2,
                            float* __restrict__ out_hidden,
                            float* __restrict__ out_logits,
                            float* __restrict__ out_v) {
    extern __shared__ unsigned char smem[];
    unsigned int* sWemb = (unsigned int*)(smem + OFFB_WEMB);
    unsigned int* sWi   = (unsigned int*)(smem + OFFB_WI);
    unsigned int* sWh   = (unsigned int*)(smem + OFFB_WH);
    unsigned int* sWa1  = (unsigned int*)(smem + OFFB_WA1);
    unsigned int* sWa2  = (unsigned int*)(smem + OFFB_WA2);
    unsigned int* sWc1  = (unsigned int*)(smem + OFFB_WC1);
    unsigned int* sWc2  = (unsigned int*)(smem + OFFB_WC2);
    float* s_bemb = (float*)(smem + OFFB_BIAS);
    float* s_bi   = s_bemb + 128;
    float* s_bhn  = s_bi + 384;
    float* s_ba1  = s_bhn + 128;
    float* s_ba2  = s_ba1 + 128;
    float* s_bc1  = s_ba2 + 16;
    float* s_bc2  = s_bc1 + 128;
    unsigned short* h_bf   = (unsigned short*)(smem + OFFB_HBF);
    unsigned short* emb_bf = (unsigned short*)(smem + OFFB_EMB);
    unsigned short* obs_bf = (unsigned short*)(smem + OFFB_OBS);
    unsigned short* a1_bf  = (unsigned short*)(smem + OFFB_A1);
    unsigned short* c1_bf  = (unsigned short*)(smem + OFFB_C1);
    unsigned short* sdall  = (unsigned short*)(smem + OFFB_DMASK);

    const int tid  = threadIdx.x;
    const int lane = tid & 31;
    const int w    = tid >> 5;                 // wave id 0..7 (wave32)
    const int B0   = blockIdx.x * BT_;         // batch-row tile base
    const v8f z8   = {0.f, 0.f, 0.f, 0.f, 0.f, 0.f, 0.f, 0.f};

    // ---- one-time staging: weights -> kpair-packed bf16 in LDS, biases, done masks ----
    stage_packed(W_emb, sWemb, OBS_, H_,     tid);
    stage_packed(Wi,    sWi,   H_,   3 * H_, tid);
    stage_packed(Wh,    sWh,   H_,   3 * H_, tid);
    stage_packed(Wa1,   sWa1,  H_,   H_,     tid);
    stage_packed(Wa2,   sWa2,  H_,   A_,     tid);
    stage_packed(Wc1,   sWc1,  H_,   H_,     tid);
    for (int e = tid; e < 64 * 16; e += NTHR) {          // Wc2 [128,1] padded to N=16
        const int kp = e >> 4, n = e & 15;
        sWc2[e] = (n == 0) ? pack2bf(Wc2[2 * kp], Wc2[2 * kp + 1]) : 0u;
    }
    for (int e = tid; e < 128; e += NTHR) { s_bemb[e] = b_emb[e]; s_bhn[e] = bh_n[e];
                                            s_ba1[e] = ba1[e];   s_bc1[e] = bc1[e]; }
    for (int e = tid; e < 384; e += NTHR) s_bi[e] = bi[e];
    for (int e = tid; e < 16;  e += NTHR) { s_ba2[e] = ba2[e]; s_bc2[e] = (e == 0) ? bc2[0] : 0.f; }
    for (int tt = tid; tt < T_; tt += NTHR) {            // done bitmasks, whole rollout
        const int* dp = dones + (size_t)tt * B_ + B0;
        unsigned m = 0;
#pragma unroll
        for (int j = 0; j < BT_; ++j) m |= (dp[j] != 0 ? 1u : 0u) << j;
        sdall[tt] = (unsigned short)m;
    }

    const int col = 16 * w + (lane & 15);      // this wave's output column slice
    const int mhi = (lane >> 4) << 3;          // C-fragment row offset for this half-wave

    // f32 carry lives in registers: hcar[i] is h[B0+i+mhi][col], owned by this lane only
    float hcar[8];
#pragma unroll
    for (int i = 0; i < 8; ++i)
        hcar[i] = hidden0[(size_t)(B0 + i + mhi) * H_ + col];
    for (int e = tid; e < BT_ * H_; e += NTHR) {
        const int m = e >> 7, c = e & 127;
        h_bf[e] = f2bf(hidden0[(size_t)(B0 + m) * H_ + c]);
    }
    __syncthreads();

#pragma unroll 1
    for (int t = 0; t < T_; ++t) {
        const unsigned dmask = sdall[t];       // LDS broadcast, pre-resident
        // P1: stage obs tile (coalesced float2 -> packed bf16); reset h_bf on done;
        //     prefetch next obs tile
        for (int e = tid; e < BT_ * OBS_ / 2; e += NTHR) {
            const int m = e >> 5, o = e & 31;
            const float2 v = *(const float2*)(&obs[((size_t)t * B_ + (B0 + m)) * OBS_ + 2 * o]);
            ((unsigned int*)obs_bf)[e] = pack2bf(v.x, v.y);
        }
        if (dmask) {
            unsigned int* hb32 = (unsigned int*)h_bf;
            for (int e = tid; e < BT_ * H_ / 2; e += NTHR)
                if ((dmask >> (e >> 6)) & 1) hb32[e] = 0u;
        }
        if (t + 1 < T_ && tid < 32) {
            const int m = tid & 15, half = (tid >> 4) & 1;
            __builtin_prefetch(&obs[((size_t)(t + 1) * B_ + (B0 + m)) * OBS_ + half * 32], 0, 0);
        }
        __syncthreads();   // B1: obs_bf + h_bf(reset) visible

        // P2: emb = relu(obs@W_emb+b_emb) (writes emb_bf), and hh = h@Wh (register accs).
        //     hh does not need emb, so one barrier below covers both dependencies.
        v8f hr = z8, hz = z8, hn = z8;
        {
            v8f acc = gemm1(obs_bf, OBS_, OBS_, sWemb, H_, 16 * w, lane, z8);
            const float be = s_bemb[col];
#pragma unroll
            for (int i = 0; i < 8; ++i)
                emb_bf[(i + mhi) * H_ + col] = f2bf(fmaxf(acc[i] + be, 0.f));
            gemm3(h_bf, H_, sWh, col, lane, hr, hz, hn);
        }
        __syncthreads();   // B2: emb_bf visible AND all h_bf reads complete (WAR)

        // P3: xi = emb@Wi (shared A tile), then per-lane gate math on matching fragments
        v8f xr = z8, xz = z8, xn = z8;
        gemm3(emb_bf, H_, sWi, col, lane, xr, xz, xn);
        const float bir = s_bi[col], biz = s_bi[col + H_], bin = s_bi[col + 2 * H_];
        const float bhn = s_bhn[col];
#pragma unroll
        for (int i = 0; i < 8; ++i) {
            const float hp = ((dmask >> (i + mhi)) & 1) ? 0.f : hcar[i];
            const float r  = sigmoidf_(xr[i] + bir + hr[i]);
            const float zz = sigmoidf_(xz[i] + biz + hz[i]);
            const float ng = tanhf(xn[i] + bin + r * (hn[i] + bhn));
            const float hv = (1.f - zz) * ng + zz * hp;
            hcar[i] = hv;
            h_bf[(i + mhi) * H_ + col] = f2bf(hv);   // y_t bf16 = next A-fragments
        }
        __syncthreads();   // B3: y_t visible

        // P4: heads layer 1 (shared A tile): a1 = relu(y@Wa1+ba1), c1 = relu(y@Wc1+bc1)
        {
            v8f a1 = z8, c1 = z8;
            gemm2(h_bf, H_, sWa1, sWc1, col, lane, a1, c1);
            const float ba = s_ba1[col], bc = s_bc1[col];
#pragma unroll
            for (int i = 0; i < 8; ++i) {
                a1_bf[(i + mhi) * H_ + col] = f2bf(fmaxf(a1[i] + ba, 0.f));
                c1_bf[(i + mhi) * H_ + col] = f2bf(fmaxf(c1[i] + bc, 0.f));
            }
        }
        __syncthreads();   // B4: a1/c1 visible

        // P5: heads layer 2 -> global
        if (w == 0) {                          // logits [16 x 16]
            const v8f lg = gemm1(a1_bf, H_, H_, sWa2, A_, 0, lane, z8);
            const int a = lane & 15;
            const float b2 = s_ba2[a];
#pragma unroll
            for (int i = 0; i < 8; ++i)
                out_logits[((size_t)t * B_ + (B0 + i + mhi)) * A_ + a] = lg[i] + b2;
        } else if (w == 1) {                   // value [16 x 1] via padded 16-col tile
            const v8f vv = gemm1(c1_bf, H_, H_, sWc2, A_, 0, lane, z8);
            if ((lane & 15) == 0) {
                const float b2 = s_bc2[0];
#pragma unroll
                for (int i = 0; i < 8; ++i)
                    out_v[(size_t)t * B_ + (B0 + i + mhi)] = vv[i] + b2;
            }
        }
        // no trailing barrier: step t+1's writes are fenced by its own B1 before any reader
    }

    // final hidden straight from register carry
#pragma unroll
    for (int i = 0; i < 8; ++i)
        out_hidden[(size_t)(B0 + i + mhi) * H_ + col] = hcar[i];
}

extern "C" void kernel_launch(void* const* d_in, const int* in_sizes, int n_in,
                              void* d_out, int out_size, void* d_ws, size_t ws_size,
                              hipStream_t stream) {
    (void)in_sizes; (void)n_in; (void)out_size; (void)d_ws; (void)ws_size;
    const float* hidden = (const float*)d_in[0];
    const float* obs    = (const float*)d_in[1];
    const int*   dones  = (const int*)  d_in[2];
    const float* W_emb  = (const float*)d_in[3];
    const float* b_emb  = (const float*)d_in[4];
    const float* Wi     = (const float*)d_in[5];
    const float* bi     = (const float*)d_in[6];
    const float* Wh     = (const float*)d_in[7];
    const float* bh_n   = (const float*)d_in[8];
    const float* Wa1    = (const float*)d_in[9];
    const float* ba1    = (const float*)d_in[10];
    const float* Wa2    = (const float*)d_in[11];
    const float* ba2    = (const float*)d_in[12];
    const float* Wc1    = (const float*)d_in[13];
    const float* bc1    = (const float*)d_in[14];
    const float* Wc2    = (const float*)d_in[15];
    const float* bc2    = (const float*)d_in[16];

    float* out        = (float*)d_out;
    float* out_hidden = out;                                  // [B,H]
    float* out_logits = out + (size_t)B_ * H_;                // [T,B,A]
    float* out_v      = out_logits + (size_t)T_ * B_ * A_;    // [T,B]

    hipFuncSetAttribute(reinterpret_cast<const void*>(actor_critic_gru_fused),
                        hipFuncAttributeMaxDynamicSharedMemorySize, (int)SMEM_BYTES);

    actor_critic_gru_fused<<<B_ / BT_, NTHR, SMEM_BYTES, stream>>>(
        hidden, obs, dones, W_emb, b_emb, Wi, bi, Wh, bh_n,
        Wa1, ba1, Wa2, ba2, Wc1, bc1, Wc2, bc2,
        out_hidden, out_logits, out_v);
}
